// Attention_21285857919287
// MI455X (gfx1250) — matrix operand order repaired and checked by reading, not compiled, not run on hardware
//
#include <hip/hip_runtime.h>

// ---------------------------------------------------------------------------
// Fused causal MHA forward for MI455X (gfx1250, wave32, WMMA bf16 path).
//   x      [B, N, Dm]   fp32
//   W_K/Q/V[H, Dh, Dm]  fp32
//   out    [B, N, H*Dh] fp32
// Pipeline:
//   0) pack fp32 -> bf16 copies of x and W (once; hardware cvt)
//   1) QKV GEMM: 32x64 tile per wave, pure bf16 loads + v_wmma_f32_16x16x32_bf16
//   2) flash attention: 16 queries per wave, online softmax, WMMA QK^T and PV
// Workspace: xb 16MB | Wb 6MB | K,Q bf16 [B,H,N,Dh] | V bf16 [B,H,Dh,N]  (70 MB)
// ---------------------------------------------------------------------------

#define BATCH   4
#define NCTX    2048
#define DMODEL  1024
#define NHEADS  16
#define DHEAD   64

typedef __attribute__((ext_vector_type(16))) __bf16 v16bf;
typedef __attribute__((ext_vector_type(8)))  float  v8f;

union Frag16 { unsigned u[8]; v16bf v; };   // 16 bf16 = 8 VGPRs (A or B fragment)

// fp32 -> bf16 through the hardware conversion path (clang __bf16 cast).
__device__ __forceinline__ unsigned short bf16bits(float f) {
    __bf16 h = (__bf16)f;
    return __builtin_bit_cast(unsigned short, h);
}
__device__ __forceinline__ unsigned pack2(float a, float b) {
    return (unsigned)bf16bits(a) | ((unsigned)bf16bits(b) << 16);
}

// A-fragment (16x32, 16-bit) dword -> K column (ISA 7.12.2):
//   lanes 0-15 kbase=0, lanes 16-31 kbase=8;  dw<4 -> kbase+2dw, dw>=4 -> 16+kbase+2(dw-4)
__device__ __forceinline__ int kOffA(int dw, int half) {
    return ((dw & 4) ? 16 : 0) + half * 8 + 2 * (dw & 3);
}
// B-fragment (32x16, 16-bit) dword -> K row: same as A: dw<4 -> 8*half+2dw, dw>=4 -> 16+8*half+2(dw-4)
__device__ __forceinline__ int kOffB(int dw, int half) {
    return ((dw & 4) ? 16 : 0) + half * 8 + 2 * (dw & 3);
}

__device__ __forceinline__ float hmax16(float v) {
    #pragma unroll
    for (int m = 1; m < 16; m <<= 1) v = fmaxf(v, __shfl_xor(v, m, 32));
    return v;
}
__device__ __forceinline__ float hsum16(float v) {
    #pragma unroll
    for (int m = 1; m < 16; m <<= 1) v += __shfl_xor(v, m, 32);
    return v;
}

// ---------------------------------------------------------------------------
// Kernel 0: packed fp32 -> bf16 conversion (2 elements / thread).
// ---------------------------------------------------------------------------
__global__ __launch_bounds__(256) void cvt_f32_bf16(
    const float* __restrict__ src, unsigned* __restrict__ dst, int npairs)
{
    const int i = blockIdx.x * blockDim.x + threadIdx.x;
    if (i < npairs) {
        const float2 f = ((const float2*)src)[i];
        dst[i] = pack2(f.x, f.y);
    }
}

// ---------------------------------------------------------------------------
// Kernel 1: QKV projection. One wave = 32 rows x 64 cols for one (b, head, w).
// All operands pre-converted bf16: inner loop is loads + WMMA only.
// grid = (NCTX/32, 3*NHEADS, BATCH), block = 32.
// ---------------------------------------------------------------------------
__global__ __launch_bounds__(32) void qkv_proj(
    const unsigned short* __restrict__ xb,   // [B, N, Dm] bf16
    const unsigned short* __restrict__ Wb,   // [3, H, Dh, Dm] bf16
    unsigned short* __restrict__ Kbuf,
    unsigned short* __restrict__ Qbuf,
    unsigned short* __restrict__ Vbuf)
{
    const int lane = threadIdx.x & 31;
    const int half = lane >> 4;
    const int l15  = lane & 15;
    const int pt   = blockIdx.x;            // 32-row tile
    const int head = blockIdx.y & (NHEADS - 1);
    const int widx = blockIdx.y >> 4;       // 0 = K, 1 = Q, 2 = V
    const int b    = blockIdx.z;

    const unsigned* xrow32[2];
    #pragma unroll
    for (int u = 0; u < 2; ++u)
        xrow32[u] = (const unsigned*)
            (xb + ((size_t)b * NCTX + pt * 32 + u * 16 + l15) * DMODEL);

    const unsigned short* Wh =
        Wb + (((size_t)widx * NHEADS + head) * DHEAD) * DMODEL;

    v8f acc[2][4];
    #pragma unroll
    for (int u = 0; u < 2; ++u)
        #pragma unroll
        for (int ct = 0; ct < 4; ++ct) acc[u][ct] = 0.0f;

    for (int kk = 0; kk < DMODEL; kk += 32) {
        Frag16 a[2];
        #pragma unroll
        for (int u = 0; u < 2; ++u)
            #pragma unroll
            for (int dw = 0; dw < 8; ++dw)
                a[u].u[dw] = xrow32[u][(kk + kOffA(dw, half)) >> 1];

        #pragma unroll
        for (int ct = 0; ct < 4; ++ct) {
            const unsigned* wrow32 =
                (const unsigned*)(Wh + (size_t)(ct * 16 + l15) * DMODEL);
            Frag16 bw;
            #pragma unroll
            for (int dw = 0; dw < 8; ++dw)
                bw.u[dw] = wrow32[(kk + kOffB(dw, half)) >> 1];
            #pragma unroll
            for (int u = 0; u < 2; ++u)     // B fragment reused by both A tiles
                acc[u][ct] = __builtin_amdgcn_wmma_f32_16x16x32_bf16(
                    false, a[u].v, false, bw.v, (short)0, acc[u][ct], false, false);
        }
    }

    // ---- epilogue: branch selected ONCE per wave, store loops branch-free --
    const size_t bh = (size_t)b * NHEADS + head;
    if (widx == 2) {
        // V stored transposed [h][p]; consecutive C rows r are consecutive p,
        // so pack pairs and emit aligned dword stores.
        unsigned short* dstV = Vbuf + bh * DHEAD * NCTX;
        #pragma unroll
        for (int u = 0; u < 2; ++u)
            #pragma unroll
            for (int ct = 0; ct < 4; ++ct) {
                const int h = ct * 16 + l15;
                unsigned* vout = (unsigned*)(dstV + (size_t)h * NCTX);
                #pragma unroll
                for (int r2 = 0; r2 < 4; ++r2) {
                    const int p0 = pt * 32 + u * 16 + 2 * r2 + half * 8;
                    vout[p0 >> 1] = pack2(acc[u][ct][2 * r2],
                                          acc[u][ct][2 * r2 + 1]);
                }
            }
    } else {
        unsigned short* dst =
            ((widx == 0) ? Kbuf : Qbuf) + bh * NCTX * DHEAD;   // [p][h]
        #pragma unroll
        for (int u = 0; u < 2; ++u)
            #pragma unroll
            for (int ct = 0; ct < 4; ++ct)
                #pragma unroll
                for (int r = 0; r < 8; ++r) {
                    const int p = pt * 32 + u * 16 + r + half * 8;
                    const int h = ct * 16 + l15;
                    dst[(size_t)p * DHEAD + h] = bf16bits(acc[u][ct][r]);
                }
    }
}

// ---------------------------------------------------------------------------
// Kernel 2: flash attention. One wave = 16 queries for one (b, head).
// grid = (NCTX/16, NHEADS, BATCH), block = 32.
// ---------------------------------------------------------------------------
__global__ __launch_bounds__(32) void attn_fwd(
    const unsigned short* __restrict__ Kbuf,
    const unsigned short* __restrict__ Qbuf,
    const unsigned short* __restrict__ Vbuf,
    float* __restrict__ out)
{
    __shared__ unsigned Plds[16 * 16];      // 16 x 32 bf16 P tile (1 KB)

    const int lane = threadIdx.x & 31;
    const int half = lane >> 4;
    const int l15  = lane & 15;
    const int qt   = blockIdx.x;
    const int head = blockIdx.y;
    const int b    = blockIdx.z;
    const int qrow = qt * 16;

    const size_t bh = (size_t)b * NHEADS + head;
    const unsigned short* __restrict__ Qb = Qbuf + bh * NCTX * DHEAD;
    const unsigned short* __restrict__ Kb = Kbuf + bh * NCTX * DHEAD;
    const unsigned short* __restrict__ Vb = Vbuf + bh * DHEAD * NCTX;

    // Q A-fragments for the two 32-wide h-chunks (row tile is fixed).
    Frag16 aQ[2];
    {
        const unsigned* qrow32 =
            (const unsigned*)(Qb + (size_t)(qrow + l15) * DHEAD);
        #pragma unroll
        for (int c = 0; c < 2; ++c)
            #pragma unroll
            for (int dw = 0; dw < 8; ++dw)
                aQ[c].u[dw] = qrow32[(c * 32 + kOffA(dw, half)) >> 1];
    }

    v8f accz[4];
    float mrow[8], lrow[8];
    #pragma unroll
    for (int ct = 0; ct < 4; ++ct) accz[ct] = 0.0f;
    #pragma unroll
    for (int r = 0; r < 8; ++r) { mrow[r] = -1e30f; lrow[r] = 0.0f; }

    const int kbmax = (qrow + 15) >> 5;     // causal: keys <= qrow+15
    for (int kb = 0; kb <= kbmax; ++kb) {
        const int pbase = kb * 32;

        // ---- scores S = (Q K^T) / sqrt(Dh), causal-masked -------------------
        float stile[2][8];
        #pragma unroll
        for (int t = 0; t < 2; ++t) {
            const int key = pbase + t * 16 + l15;    // this lane's key column
            const unsigned* krow32 =
                (const unsigned*)(Kb + (size_t)key * DHEAD);
            v8f s = 0.0f;
            #pragma unroll
            for (int c = 0; c < 2; ++c) {
                Frag16 bk;
                #pragma unroll
                for (int dw = 0; dw < 8; ++dw)
                    bk.u[dw] = krow32[(c * 32 + kOffB(dw, half)) >> 1];
                s = __builtin_amdgcn_wmma_f32_16x16x32_bf16(
                    false, aQ[c].v, false, bk.v, (short)0, s, false, false);
            }
            #pragma unroll
            for (int r = 0; r < 8; ++r) {
                const int q = qrow + r + half * 8;
                float sv = s[r] * 0.125f;            // 1/sqrt(64)
                if (key > q) sv = -1e30f;            // causal mask
                stile[t][r] = sv;
            }
        }

        // ---- online softmax update -----------------------------------------
        float mnew[8], rsum[8], pvals[2][8];
        #pragma unroll
        for (int r = 0; r < 8; ++r) {
            float v = fmaxf(stile[0][r], stile[1][r]);
            mnew[r] = fmaxf(mrow[r], hmax16(v));
        }
        #pragma unroll
        for (int r = 0; r < 8; ++r) {
            const float p0 = __expf(stile[0][r] - mnew[r]);
            const float p1 = __expf(stile[1][r] - mnew[r]);
            pvals[0][r] = p0; pvals[1][r] = p1;
            rsum[r] = hsum16(p0 + p1);
        }
        #pragma unroll
        for (int r = 0; r < 8; ++r) {
            const float alpha = __expf(mrow[r] - mnew[r]);
            lrow[r] = lrow[r] * alpha + rsum[r];
            mrow[r] = mnew[r];
            #pragma unroll
            for (int ct = 0; ct < 4; ++ct) accz[ct][r] *= alpha;
        }

        // ---- P: C-fragment -> LDS -> A-fragment (transpose round trip) -----
        __syncthreads();
        #pragma unroll
        for (int t = 0; t < 2; ++t)
            #pragma unroll
            for (int r = 0; r < 8; ++r) {
                const int m   = r + half * 8;
                const int col = t * 16 + l15;
                ((unsigned short*)Plds)[m * 32 + col] = bf16bits(pvals[t][r]);
            }
        __syncthreads();
        Frag16 aP;
        {
            const unsigned* prow = Plds + l15 * 16;  // row stride = 16 dwords
            #pragma unroll
            for (int dw = 0; dw < 8; ++dw)
                aP.u[dw] = prow[kOffA(dw, half) >> 1];
        }

        // ---- z += P V  (V stored [h][p], contiguous K gathers) --------------
        #pragma unroll
        for (int ct = 0; ct < 4; ++ct) {
            const int h = ct * 16 + l15;
            const unsigned* vrow = (const unsigned*)(Vb + (size_t)h * NCTX);
            Frag16 bv;
            #pragma unroll
            for (int dw = 0; dw < 8; ++dw)
                bv.u[dw] = vrow[(pbase + kOffB(dw, half)) >> 1];
            accz[ct] = __builtin_amdgcn_wmma_f32_16x16x32_bf16(
                false, aP.v, false, bv.v, (short)0, accz[ct], false, false);
        }
    }

    // ---- epilogue: out[b, q, head*64 + h] = z / l --------------------------
    #pragma unroll
    for (int ct = 0; ct < 4; ++ct)
        #pragma unroll
        for (int r = 0; r < 8; ++r) {
            const int q = qrow + r + half * 8;
            const int h = ct * 16 + l15;
            out[((size_t)b * NCTX + q) * (NHEADS * DHEAD) + head * DHEAD + h] =
                accz[ct][r] / lrow[r];
        }
}

// ---------------------------------------------------------------------------
extern "C" void kernel_launch(void* const* d_in, const int* in_sizes, int n_in,
                              void* d_out, int out_size, void* d_ws, size_t ws_size,
                              hipStream_t stream) {
    (void)in_sizes; (void)n_in; (void)out_size; (void)ws_size;
    const float* x  = (const float*)d_in[0];
    const float* WK = (const float*)d_in[1];
    const float* WQ = (const float*)d_in[2];
    const float* WV = (const float*)d_in[3];

    const size_t SX = (size_t)BATCH * NCTX * DMODEL;            // 8 Mi bf16
    const size_t SW = (size_t)NHEADS * DHEAD * DMODEL;          // 1 Mi bf16 per W
    const size_t SK = (size_t)BATCH * NHEADS * NCTX * DHEAD;    // 8 Mi bf16

    unsigned short* xb   = (unsigned short*)d_ws;
    unsigned short* Wb   = xb + SX;                             // [3][H][Dh][Dm]
    unsigned short* Kbuf = Wb + 3 * SW;
    unsigned short* Qbuf = Kbuf + SK;
    unsigned short* Vbuf = Qbuf + SK;                           // total ~70 MB

    // Stage 0: pack fp32 -> bf16 once.
    {
        const int px = (int)(SX / 2), pw = (int)(SW / 2);
        cvt_f32_bf16<<<(px + 255) / 256, 256, 0, stream>>>(x,  (unsigned*)xb, px);
        cvt_f32_bf16<<<(pw + 255) / 256, 256, 0, stream>>>(WK, (unsigned*)(Wb + 0 * SW), pw);
        cvt_f32_bf16<<<(pw + 255) / 256, 256, 0, stream>>>(WQ, (unsigned*)(Wb + 1 * SW), pw);
        cvt_f32_bf16<<<(pw + 255) / 256, 256, 0, stream>>>(WV, (unsigned*)(Wb + 2 * SW), pw);
    }

    // Stage 1: QKV projection.
    dim3 g1(NCTX / 32, 3 * NHEADS, BATCH);
    qkv_proj<<<g1, 32, 0, stream>>>(xb, Wb, Kbuf, Qbuf, Vbuf);

    // Stage 2: flash attention.
    dim3 g2(NCTX / 16, NHEADS, BATCH);
    attn_fwd<<<g2, 32, 0, stream>>>(Kbuf, Qbuf, Vbuf, (float*)d_out);
}
